// MoELayer_33182917329236
// MI455X (gfx1250) — matrix operand order repaired and checked by reading, not compiled, not run on hardware
//
#include <hip/hip_runtime.h>

#define TTOK  4096
#define DM    1024
#define DFF   2816
#define DSFF  1408
#define NEXP  8
#define TOPK  2

typedef __attribute__((ext_vector_type(16))) _Float16 v16h;
typedef __attribute__((ext_vector_type(8)))  _Float16 v8h;
typedef __attribute__((ext_vector_type(8)))  float    v8f;

static __device__ __forceinline__ v8f wmma_f16(v16h a, v16h b, v8f c) {
  return __builtin_amdgcn_wmma_f32_16x16x32_f16(false, a, false, b, (short)0, c,
                                                false, false);
}

static __device__ __forceinline__ v16h hzero16() {
  v16h f;
#pragma unroll
  for (int i = 0; i < 16; ++i) f[i] = (_Float16)0.f;
  return f;
}

static __device__ __forceinline__ v16h cvt16(float4 a, float4 b, float4 c, float4 d) {
  v16h f;
  f[0]=(_Float16)a.x;  f[1]=(_Float16)a.y;  f[2]=(_Float16)a.z;  f[3]=(_Float16)a.w;
  f[4]=(_Float16)b.x;  f[5]=(_Float16)b.y;  f[6]=(_Float16)b.z;  f[7]=(_Float16)b.w;
  f[8]=(_Float16)c.x;  f[9]=(_Float16)c.y;  f[10]=(_Float16)c.z; f[11]=(_Float16)c.w;
  f[12]=(_Float16)d.x; f[13]=(_Float16)d.y; f[14]=(_Float16)d.z; f[15]=(_Float16)d.w;
  return f;
}

// ---- A fragment (16 rows x 32 K): lane holds row
// lane&15; 16 contiguous K starting at kb+16g.
static __device__ __forceinline__ v16h load_a(const float* rowptr, int kb, int g) {
  if (!rowptr) return hzero16();
  const float4* p0 = reinterpret_cast<const float4*>(rowptr + kb + 16 * g);
  return cvt16(p0[0], p0[1], p0[2], p0[3]);
}
static __device__ __forceinline__ v16h load_a(const _Float16* rowptr, int kb, int g) {
  if (!rowptr) return hzero16();
  union { v16h v; v8h h[2]; } u;
  u.h[0] = *reinterpret_cast<const v8h*>(rowptr + kb + 16 * g);
  u.h[1] = *reinterpret_cast<const v8h*>(rowptr + kb + 16 * g + 8);
  return u.v;
}
// ---- B fragment (32 K x 16 cols) for X @ W^T with W row-major [N,K]:
// lane holds column n = lane&15, 16 contiguous K starting at kb+16g.
static __device__ __forceinline__ v16h load_b(const float* wrow, int kb, int g) {
  const float4* p = reinterpret_cast<const float4*>(wrow + kb + 16 * g);
  return cvt16(p[0], p[1], p[2], p[3]);
}
static __device__ __forceinline__ v16h load_b(const _Float16* wrow, int kb, int g) {
  union { v16h v; v8h h[2]; } u;
  const _Float16* p = wrow + kb + 16 * g;
  u.h[0] = *reinterpret_cast<const v8h*>(p);
  u.h[1] = *reinterpret_cast<const v8h*>(p + 8);
  return u.v;
}

// ---- CDNA5 async global->LDS (ASYNCcnt) --------------------------------
static __device__ __forceinline__ void async_ld16(unsigned lds_byte_off,
                                                  const void* gptr) {
  asm volatile("global_load_async_to_lds_b128 %0, %1, off"
               :: "v"(lds_byte_off), "v"(gptr) : "memory");
}
#define WAIT_ASYNCCNT(n) asm volatile("s_wait_asynccnt " #n ::: "memory")
// generic-pointer low 32 bits == LDS byte offset (ISA flat-aperture rule)
#define LDS_OFF(p) ((unsigned)(unsigned long long)(const void*)(p))

// ---------------------------------------------------------------- f32->f16 ---
__global__ __launch_bounds__(256) void cvt_kernel(const float* __restrict__ src,
                                                  _Float16* __restrict__ dst,
                                                  size_t n) {
  size_t i = ((size_t)blockIdx.x * blockDim.x + threadIdx.x) * 8;
  if (i >= n) return;
  float4 a = *reinterpret_cast<const float4*>(src + i);
  float4 b = *reinterpret_cast<const float4*>(src + i + 4);
  v8h o;
  o[0]=(_Float16)a.x; o[1]=(_Float16)a.y; o[2]=(_Float16)a.z; o[3]=(_Float16)a.w;
  o[4]=(_Float16)b.x; o[5]=(_Float16)b.y; o[6]=(_Float16)b.z; o[7]=(_Float16)b.w;
  *reinterpret_cast<v8h*>(dst + i) = o;
}

// ---------------------------------------------------------------- router ---
__global__ __launch_bounds__(256) void router_kernel(
    const float* __restrict__ x, const float* __restrict__ gw,
    int* __restrict__ topi, float* __restrict__ topw,
    float* __restrict__ lse2, float* __restrict__ probs) {
  const int t = blockIdx.x * 8 + (threadIdx.x >> 5);
  const int lane = threadIdx.x & 31;
  const float* xr = x + (size_t)t * DM;
  float acc[NEXP];
#pragma unroll
  for (int e = 0; e < NEXP; ++e) acc[e] = 0.f;
  for (int d = lane; d < DM; d += 32) {
    float xv = xr[d];
#pragma unroll
    for (int e = 0; e < NEXP; ++e) acc[e] += xv * gw[e * DM + d];
  }
#pragma unroll
  for (int e = 0; e < NEXP; ++e)
#pragma unroll
    for (int s = 16; s > 0; s >>= 1) acc[e] += __shfl_xor(acc[e], s, 32);
  if (lane == 0) {
    float mx = acc[0];
#pragma unroll
    for (int e = 1; e < NEXP; ++e) mx = fmaxf(mx, acc[e]);
    float ex[NEXP], sum = 0.f;
#pragma unroll
    for (int e = 0; e < NEXP; ++e) { ex[e] = __expf(acc[e] - mx); sum += ex[e]; }
    float lse = mx + __logf(sum);
    lse2[t] = lse * lse;
    float isum = 1.f / sum;
    float p[NEXP];
#pragma unroll
    for (int e = 0; e < NEXP; ++e) { p[e] = ex[e] * isum; probs[e * TTOK + t] = p[e]; }
    int i0 = 0; float v0 = p[0];
#pragma unroll
    for (int e = 1; e < NEXP; ++e) if (p[e] > v0) { v0 = p[e]; i0 = e; }
    int i1 = -1; float v1 = -1.f;
#pragma unroll
    for (int e = 0; e < NEXP; ++e)
      if (e != i0 && p[e] > v1) { v1 = p[e]; i1 = e; }
    float sw = v0 + v1;
    topi[2 * t] = i0;      topi[2 * t + 1] = i1;
    topw[2 * t] = v0 / sw; topw[2 * t + 1] = v1 / sw;
  }
}

// ------------------------------------------------- losses + segment offsets ---
__global__ __launch_bounds__(256) void finalize_kernel(
    const int* __restrict__ topi, const float* __restrict__ lse2,
    const float* __restrict__ probs,
    int* __restrict__ off, int* __restrict__ ecnt,
    float* __restrict__ out_tail) {
  __shared__ float redf[256];
  __shared__ int   redi[256];
  __shared__ float s_z;
  __shared__ float s_ps[NEXP];
  __shared__ int   s_cn[NEXP];
  const int tid = threadIdx.x;
  float s = 0.f;
  for (int t = tid; t < TTOK; t += 256) s += lse2[t];
  redf[tid] = s; __syncthreads();
  for (int st = 128; st > 0; st >>= 1) {
    if (tid < st) redf[tid] += redf[tid + st];
    __syncthreads();
  }
  if (tid == 0) s_z = redf[0];
  __syncthreads();
  for (int e = 0; e < NEXP; ++e) {
    float ps = 0.f; int c = 0;
    for (int t = tid; t < TTOK; t += 256) {
      ps += probs[e * TTOK + t];
      c  += (topi[2 * t] == e) + (topi[2 * t + 1] == e);
    }
    redf[tid] = ps; redi[tid] = c; __syncthreads();
    for (int st = 128; st > 0; st >>= 1) {
      if (tid < st) { redf[tid] += redf[tid + st]; redi[tid] += redi[tid + st]; }
      __syncthreads();
    }
    if (tid == 0) { s_ps[e] = redf[0]; s_cn[e] = redi[0]; }
    __syncthreads();
  }
  if (tid == 0) {
    int acc = 0;
    for (int e = 0; e < NEXP; ++e) {
      off[e] = acc; acc += s_cn[e]; ecnt[e] = s_cn[e];
      out_tail[1 + e] = (float)s_cn[e];
    }
    off[NEXP] = acc;
    float bal = 0.f;
    for (int e = 0; e < NEXP; ++e)
      bal += (s_ps[e] / (float)TTOK) * ((float)s_cn[e] / (float)(TTOK * TOPK));
    out_tail[0] = 0.001f * (s_z / (float)TTOK) + 0.01f * (float)NEXP * bal;
  }
}

// ---------------------------- deterministic per-expert token-list compaction ---
__global__ __launch_bounds__(256) void pack_kernel(
    const int* __restrict__ topi, const int* __restrict__ off,
    int* __restrict__ ptok, int* __restrict__ inv) {
  const int e = blockIdx.x;
  const int tid = threadIdx.x;
  __shared__ int buf[256];
  const int t0 = tid * 16;
  int cntl = 0;
#pragma unroll
  for (int i = 0; i < 16; ++i) {
    int t = t0 + i;
    cntl += (topi[2 * t] == e) + (topi[2 * t + 1] == e);
  }
  buf[tid] = cntl; __syncthreads();
  for (int s = 1; s < 256; s <<= 1) {
    int v = (tid >= s) ? buf[tid - s] : 0;
    __syncthreads();
    buf[tid] += v;
    __syncthreads();
  }
  int base = off[e] + (buf[tid] - cntl);
#pragma unroll
  for (int i = 0; i < 16; ++i) {
    int t = t0 + i;
    if (topi[2 * t]     == e) { ptok[base] = t; inv[2 * t]     = base; ++base; }
    if (topi[2 * t + 1] == e) { ptok[base] = t; inv[2 * t + 1] = base; ++base; }
  }
}

// ---------------- SwiGLU up, direct f32 path (fallback when ws is small) -----
template <bool ROUTED>
__global__ __launch_bounds__(256) void swiglu_up_kernel(
    const float* __restrict__ x, const float* __restrict__ w1,
    const float* __restrict__ w3, const int* __restrict__ ptok,
    const int* __restrict__ off, const int* __restrict__ ecnt,
    _Float16* __restrict__ H, int Fdim) {
  const int e    = ROUTED ? blockIdx.z : 0;
  const int rows = ROUTED ? ecnt[e] : TTOK;
  const int rb   = ROUTED ? off[e] : 0;
  const int lane = threadIdx.x & 31;
  const int wv   = threadIdx.x >> 5;
  const int M0   = blockIdx.x * 128 + (wv >> 1) * 32;
  const int N0   = blockIdx.y * 64 + (wv & 1) * 32;
  if (M0 >= rows) return;
  const int g = lane >> 4, ln = lane & 15;
  const float* W1 = w1 + (size_t)e * Fdim * DM;
  const float* W3 = w3 + (size_t)e * Fdim * DM;
  const float* arow[2];
#pragma unroll
  for (int i = 0; i < 2; ++i) {
    int m = M0 + 16 * i + ln;
    if (m < rows) {
      int tok = ROUTED ? ptok[rb + m] : m;
      arow[i] = x + (size_t)tok * DM;
    } else arow[i] = nullptr;
  }
  const float* bg[2]; const float* bu[2];
#pragma unroll
  for (int j = 0; j < 2; ++j) {
    int n = N0 + 16 * j + ln;
    bg[j] = W1 + (size_t)n * DM;
    bu[j] = W3 + (size_t)n * DM;
  }
  v8f cg[2][2] = {}, cu[2][2] = {};
  for (int kb = 0; kb < DM; kb += 32) {
    v16h a0 = load_a(arow[0], kb, g);
    v16h a1 = load_a(arow[1], kb, g);
    v16h g0 = load_b(bg[0], kb, g);
    v16h g1 = load_b(bg[1], kb, g);
    v16h u0 = load_b(bu[0], kb, g);
    v16h u1 = load_b(bu[1], kb, g);
    cg[0][0] = wmma_f16(a0, g0, cg[0][0]);
    cg[0][1] = wmma_f16(a0, g1, cg[0][1]);
    cg[1][0] = wmma_f16(a1, g0, cg[1][0]);
    cg[1][1] = wmma_f16(a1, g1, cg[1][1]);
    cu[0][0] = wmma_f16(a0, u0, cu[0][0]);
    cu[0][1] = wmma_f16(a0, u1, cu[0][1]);
    cu[1][0] = wmma_f16(a1, u0, cu[1][0]);
    cu[1][1] = wmma_f16(a1, u1, cu[1][1]);
  }
#pragma unroll
  for (int i = 0; i < 2; ++i)
#pragma unroll
    for (int j = 0; j < 2; ++j) {
      int n = N0 + 16 * j + ln;
#pragma unroll
      for (int r = 0; r < 8; ++r) {
        int m = M0 + 16 * i + g * 8 + r;
        if (m < rows) {
          float gv = cg[i][j][r];
          float hv = (gv / (1.f + __expf(-gv))) * cu[i][j][r];
          H[(size_t)(rb + m) * Fdim + n] = (_Float16)hv;
        }
      }
    }
}

// ------- SwiGLU up, f16 operands staged through LDS via async copies --------
template <bool ROUTED>
__global__ __launch_bounds__(256) void swiglu_up_staged(
    const _Float16* __restrict__ x, const _Float16* __restrict__ w1,
    const _Float16* __restrict__ w3, const int* __restrict__ ptok,
    const int* __restrict__ off, const int* __restrict__ ecnt,
    _Float16* __restrict__ H, int Fdim) {
  constexpr int RSTR = 40;            // padded row stride in halves (80 B)
  constexpr int BUFH = 256 * RSTR;    // A:128 rows + Bg:64 + Bu:64
  __shared__ __align__(16) _Float16 smem[2 * BUFH];
  const int e    = ROUTED ? blockIdx.z : 0;
  const int rows = ROUTED ? ecnt[e] : TTOK;
  const int rb   = ROUTED ? off[e] : 0;
  const int M0b  = blockIdx.x * 128;
  const int N0b  = blockIdx.y * 64;
  if (M0b >= rows) return;
  const int tid = threadIdx.x;
  const _Float16* W1 = w1 + (size_t)e * Fdim * DM;
  const _Float16* W3 = w3 + (size_t)e * Fdim * DM;

  // ---- per-thread fill descriptors: 2 A chunks + 1 Bg + 1 Bu (16 B each)
  const _Float16* gA[2]; unsigned lA[2];
#pragma unroll
  for (int q = 0; q < 2; ++q) {
    int ca = tid + 256 * q;
    int r = ca >> 2, c = ca & 3;
    int m = M0b + r; if (m >= rows) m = rows - 1;   // clamp, garbage rows unused
    int tok = ROUTED ? ptok[rb + m] : m;
    gA[q] = x + (size_t)tok * DM + c * 8;
    lA[q] = LDS_OFF(&smem[r * RSTR]) + c * 16;
  }
  const int rB = tid >> 2, cB = tid & 3;
  const _Float16* gBg = W1 + (size_t)(N0b + rB) * DM + cB * 8;
  const _Float16* gBu = W3 + (size_t)(N0b + rB) * DM + cB * 8;
  const unsigned lBg = LDS_OFF(&smem[(128 + rB) * RSTR]) + cB * 16;
  const unsigned lBu = LDS_OFF(&smem[(192 + rB) * RSTR]) + cB * 16;

  // ---- per-lane consume pointers (WMMA fragment layout inside LDS tiles)
  const int lane = tid & 31, wv = tid >> 5;
  const int wm = wv >> 1, wn = wv & 1;
  const int g = lane >> 4, ln = lane & 15;
  const _Float16* aP[2]; const _Float16* bgP[2]; const _Float16* buP[2];
#pragma unroll
  for (int i = 0; i < 2; ++i)
    aP[i]  = &smem[(wm * 32 + 16 * i + ln) * RSTR + 16 * g];
#pragma unroll
  for (int j = 0; j < 2; ++j) {
    bgP[j] = &smem[(128 + wn * 32 + 16 * j + ln) * RSTR + 16 * g];
    buP[j] = &smem[(192 + wn * 32 + 16 * j + ln) * RSTR + 16 * g];
  }

  auto fill = [&](int b, int kb) {
    unsigned bo = (unsigned)(b * BUFH * 2);
#pragma unroll
    for (int q = 0; q < 2; ++q) async_ld16(lA[q] + bo, gA[q] + kb);
    async_ld16(lBg + bo, gBg + kb);
    async_ld16(lBu + bo, gBu + kb);
  };

  v8f cg[2][2] = {}, cu[2][2] = {};
  fill(0, 0);
  const int NIT = DM / 32;
  for (int it = 0; it < NIT; ++it) {
    if (it + 1 < NIT) {
      fill((it + 1) & 1, (it + 1) * 32);
      WAIT_ASYNCCNT(4);                 // previous buffer's 4 copies done
    } else {
      WAIT_ASYNCCNT(0);
    }
    __syncthreads();
    const int bo = (it & 1) * BUFH;
    union U { v16h v; v8h h[2]; };
    U a0, a1, g0, g1, u0, u1;
    a0.h[0] = *(const v8h*)(aP[0] + bo);      a0.h[1] = *(const v8h*)(aP[0] + bo + 8);
    a1.h[0] = *(const v8h*)(aP[1] + bo);      a1.h[1] = *(const v8h*)(aP[1] + bo + 8);
    g0.h[0] = *(const v8h*)(bgP[0] + bo);     g0.h[1] = *(const v8h*)(bgP[0] + bo + 8);
    g1.h[0] = *(const v8h*)(bgP[1] + bo);     g1.h[1] = *(const v8h*)(bgP[1] + bo + 8);
    u0.h[0] = *(const v8h*)(buP[0] + bo);     u0.h[1] = *(const v8h*)(buP[0] + bo + 8);
    u1.h[0] = *(const v8h*)(buP[1] + bo);     u1.h[1] = *(const v8h*)(buP[1] + bo + 8);
    cg[0][0] = wmma_f16(a0.v, g0.v, cg[0][0]);
    cg[0][1] = wmma_f16(a0.v, g1.v, cg[0][1]);
    cg[1][0] = wmma_f16(a1.v, g0.v, cg[1][0]);
    cg[1][1] = wmma_f16(a1.v, g1.v, cg[1][1]);
    cu[0][0] = wmma_f16(a0.v, u0.v, cu[0][0]);
    cu[0][1] = wmma_f16(a0.v, u1.v, cu[0][1]);
    cu[1][0] = wmma_f16(a1.v, u0.v, cu[1][0]);
    cu[1][1] = wmma_f16(a1.v, u1.v, cu[1][1]);
    __syncthreads();
  }
  const int M0 = M0b + wm * 32;
  const int N0 = N0b + wn * 32;
#pragma unroll
  for (int i = 0; i < 2; ++i)
#pragma unroll
    for (int j = 0; j < 2; ++j) {
      int n = N0 + 16 * j + ln;
#pragma unroll
      for (int r = 0; r < 8; ++r) {
        int m = M0 + 16 * i + g * 8 + r;
        if (m < rows) {
          float gv = cg[i][j][r];
          float hv = (gv / (1.f + __expf(-gv))) * cu[i][j][r];
          H[(size_t)(rb + m) * Fdim + n] = (_Float16)hv;
        }
      }
    }
}

// ------------------------------------------------- down: Out = H @ W2^T ---
template <typename TB, bool ROUTED>
__global__ __launch_bounds__(256) void down_kernel(
    const _Float16* __restrict__ H, const TB* __restrict__ w2,
    const int* __restrict__ off, const int* __restrict__ ecnt,
    float* __restrict__ out, int Kdim) {
  const int e    = ROUTED ? blockIdx.z : 0;
  const int rows = ROUTED ? ecnt[e] : TTOK;
  const int rb   = ROUTED ? off[e] : 0;
  const int lane = threadIdx.x & 31;
  const int wv   = threadIdx.x >> 5;
  const int M0   = blockIdx.x * 128 + (wv >> 1) * 32;
  const int N0   = blockIdx.y * 64 + (wv & 1) * 32;
  if (M0 >= rows) return;
  const int g = lane >> 4, ln = lane & 15;
  const TB* W2 = w2 + (ROUTED ? (size_t)e * DM * Kdim : 0);
  const _Float16* arow[2];
#pragma unroll
  for (int i = 0; i < 2; ++i) {
    int m = M0 + 16 * i + ln;
    arow[i] = (m < rows) ? H + (size_t)(rb + m) * Kdim : nullptr;
  }
  const TB* brow[2];
#pragma unroll
  for (int j = 0; j < 2; ++j) brow[j] = W2 + (size_t)(N0 + 16 * j + ln) * Kdim;
  v8f c[2][2] = {};
  for (int kb = 0; kb < Kdim; kb += 32) {
    v16h a0 = load_a(arow[0], kb, g);
    v16h a1 = load_a(arow[1], kb, g);
    v16h b0 = load_b(brow[0], kb, g);
    v16h b1 = load_b(brow[1], kb, g);
    c[0][0] = wmma_f16(a0, b0, c[0][0]);
    c[0][1] = wmma_f16(a0, b1, c[0][1]);
    c[1][0] = wmma_f16(a1, b0, c[1][0]);
    c[1][1] = wmma_f16(a1, b1, c[1][1]);
  }
#pragma unroll
  for (int i = 0; i < 2; ++i)
#pragma unroll
    for (int j = 0; j < 2; ++j) {
      int n = N0 + 16 * j + ln;
#pragma unroll
      for (int r = 0; r < 8; ++r) {
        int m = M0 + 16 * i + g * 8 + r;
        if (m < rows) out[(size_t)(rb + m) * DM + n] = c[i][j][r];
      }
    }
}

// ----------------------- out += w0*Dbuf[inv0] + w1*Dbuf[inv1] ---------------
__global__ __launch_bounds__(256) void combine_kernel(
    const float* __restrict__ Dbuf, const int* __restrict__ inv,
    const float* __restrict__ topw, float* __restrict__ out) {
  int idx = blockIdx.x * blockDim.x + threadIdx.x;
  int t = idx >> 8;
  int dq = idx & 255;
  int r0 = inv[2 * t], r1 = inv[2 * t + 1];
  float w0 = topw[2 * t], w1 = topw[2 * t + 1];
  float4 a = reinterpret_cast<const float4*>(Dbuf + (size_t)r0 * DM)[dq];
  float4 b = reinterpret_cast<const float4*>(Dbuf + (size_t)r1 * DM)[dq];
  float4* po = reinterpret_cast<float4*>(out + (size_t)t * DM) + dq;
  float4 o = *po;
  o.x += w0 * a.x + w1 * b.x;
  o.y += w0 * a.y + w1 * b.y;
  o.z += w0 * a.z + w1 * b.z;
  o.w += w0 * a.w + w1 * b.w;
  *po = o;
}

// ------------------------------------------------------------ workspace map ---
static constexpr size_t WS_TOPI = 0;
static constexpr size_t WS_TOPW = WS_TOPI + (size_t)2 * TTOK * 4;
static constexpr size_t WS_LSE2 = WS_TOPW + (size_t)2 * TTOK * 4;
static constexpr size_t WS_PROB = WS_LSE2 + (size_t)TTOK * 4;
static constexpr size_t WS_OFF  = WS_PROB + (size_t)NEXP * TTOK * 4;
static constexpr size_t WS_ECNT = WS_OFF + 256;
static constexpr size_t WS_PTOK = WS_ECNT + 256;
static constexpr size_t WS_INV  = WS_PTOK + (size_t)2 * TTOK * 4;
static constexpr size_t WS_HS   = (WS_INV + (size_t)2 * TTOK * 4 + 255) & ~(size_t)255;
static constexpr size_t WS_HR   = WS_HS + (size_t)TTOK * DSFF * 2;
static constexpr size_t WS_DB   = WS_HR + (size_t)2 * TTOK * DFF * 2;
static constexpr size_t WS_BASE_END = WS_DB + (size_t)2 * TTOK * DM * 4;
// f16 pre-converted operands (optional region)
static constexpr size_t WS_X16  = WS_BASE_END;
static constexpr size_t WS_W116 = WS_X16  + (size_t)TTOK * DM * 2;
static constexpr size_t WS_W316 = WS_W116 + (size_t)NEXP * DFF * DM * 2;
static constexpr size_t WS_W216 = WS_W316 + (size_t)NEXP * DFF * DM * 2;
static constexpr size_t WS_S116 = WS_W216 + (size_t)NEXP * DFF * DM * 2;
static constexpr size_t WS_S316 = WS_S116 + (size_t)DSFF * DM * 2;
static constexpr size_t WS_S216 = WS_S316 + (size_t)DSFF * DM * 2;
static constexpr size_t WS_F16_END = WS_S216 + (size_t)DM * DSFF * 2;

extern "C" void kernel_launch(void* const* d_in, const int* in_sizes, int n_in,
                              void* d_out, int out_size, void* d_ws, size_t ws_size,
                              hipStream_t stream) {
  const float* x      = (const float*)d_in[0];
  const float* gate_w = (const float*)d_in[1];
  const float* w1     = (const float*)d_in[2];
  const float* w2     = (const float*)d_in[3];
  const float* w3     = (const float*)d_in[4];
  const float* sw1    = (const float*)d_in[5];
  const float* sw2    = (const float*)d_in[6];
  const float* sw3    = (const float*)d_in[7];
  float* out = (float*)d_out;
  char* ws = (char*)d_ws;

  int*      topi = (int*)(ws + WS_TOPI);
  float*    topw = (float*)(ws + WS_TOPW);
  float*    lse2 = (float*)(ws + WS_LSE2);
  float*    prob = (float*)(ws + WS_PROB);
  int*      off  = (int*)(ws + WS_OFF);
  int*      ecnt = (int*)(ws + WS_ECNT);
  int*      ptok = (int*)(ws + WS_PTOK);
  int*      inv  = (int*)(ws + WS_INV);
  _Float16* Hs   = (_Float16*)(ws + WS_HS);
  _Float16* Hr   = (_Float16*)(ws + WS_HR);
  float*    Dbuf = (float*)(ws + WS_DB);

  router_kernel<<<TTOK / 8, 256, 0, stream>>>(x, gate_w, topi, topw, lse2, prob);
  finalize_kernel<<<1, 256, 0, stream>>>(topi, lse2, prob, off, ecnt,
                                         out + (size_t)TTOK * DM);
  pack_kernel<<<NEXP, 256, 0, stream>>>(topi, off, ptok, inv);

  if (ws_size >= WS_F16_END) {
    _Float16* X16  = (_Float16*)(ws + WS_X16);
    _Float16* W116 = (_Float16*)(ws + WS_W116);
    _Float16* W316 = (_Float16*)(ws + WS_W316);
    _Float16* W216 = (_Float16*)(ws + WS_W216);
    _Float16* S116 = (_Float16*)(ws + WS_S116);
    _Float16* S316 = (_Float16*)(ws + WS_S316);
    _Float16* S216 = (_Float16*)(ws + WS_S216);
    const size_t NX = (size_t)TTOK * DM, NW = (size_t)NEXP * DFF * DM,
                 NS = (size_t)DSFF * DM;
    cvt_kernel<<<(unsigned)(NX / 2048), 256, 0, stream>>>(x,   X16,  NX);
    cvt_kernel<<<(unsigned)(NW / 2048), 256, 0, stream>>>(w1,  W116, NW);
    cvt_kernel<<<(unsigned)(NW / 2048), 256, 0, stream>>>(w3,  W316, NW);
    cvt_kernel<<<(unsigned)(NW / 2048), 256, 0, stream>>>(w2,  W216, NW);
    cvt_kernel<<<(unsigned)(NS / 2048), 256, 0, stream>>>(sw1, S116, NS);
    cvt_kernel<<<(unsigned)(NS / 2048), 256, 0, stream>>>(sw3, S316, NS);
    cvt_kernel<<<(unsigned)(NS / 2048), 256, 0, stream>>>(sw2, S216, NS);
    swiglu_up_staged<false><<<dim3(TTOK / 128, DSFF / 64, 1), 256, 0, stream>>>(
        X16, S116, S316, nullptr, nullptr, nullptr, Hs, DSFF);
    down_kernel<_Float16, false><<<dim3(TTOK / 128, DM / 64, 1), 256, 0, stream>>>(
        Hs, S216, nullptr, nullptr, out, DSFF);
    swiglu_up_staged<true><<<dim3(TTOK / 128, DFF / 64, NEXP), 256, 0, stream>>>(
        X16, W116, W316, ptok, off, ecnt, Hr, DFF);
    down_kernel<_Float16, true><<<dim3(TTOK / 128, DM / 64, NEXP), 256, 0, stream>>>(
        Hr, W216, off, ecnt, Dbuf, DFF);
  } else {
    swiglu_up_kernel<false><<<dim3(TTOK / 128, DSFF / 64, 1), 256, 0, stream>>>(
        x, sw1, sw3, nullptr, nullptr, nullptr, Hs, DSFF);
    down_kernel<float, false><<<dim3(TTOK / 128, DM / 64, 1), 256, 0, stream>>>(
        Hs, sw2, nullptr, nullptr, out, DSFF);
    swiglu_up_kernel<true><<<dim3(TTOK / 128, DFF / 64, NEXP), 256, 0, stream>>>(
        x, w1, w3, ptok, off, ecnt, Hr, DFF);
    down_kernel<float, true><<<dim3(TTOK / 128, DM / 64, NEXP), 256, 0, stream>>>(
        Hr, w2, off, ecnt, Dbuf, DFF);
  }
  combine_kernel<<<(TTOK * (DM / 4)) / 256, 256, 0, stream>>>(Dbuf, inv, topw, out);
}
